// CustomLoss_14422500180565
// MI455X (gfx1250) — compile-verified, matrix operand order checked by reading, not run on hardware
//
#include <hip/hip_runtime.h>

typedef __attribute__((ext_vector_type(16))) _Float16 v16h;
typedef __attribute__((ext_vector_type(8)))  float    v8f;

#define IMH 4096
#define IMW 4096
#define TG22F 0.4142135623730951f

// ---------------------------------------------------------------------------
// Kernel 0: zero the MSE accumulator
// ---------------------------------------------------------------------------
__global__ void k_zero(float* acc) {
    if (threadIdx.x == 0 && blockIdx.x == 0) *acc = 0.0f;
}

// ---------------------------------------------------------------------------
// Kernel 1: MSE partial sums (grid-stride, wave32 shuffle reduce, atomicAdd)
// ---------------------------------------------------------------------------
__global__ __launch_bounds__(256) void k_mse(const float* __restrict__ x,
                                             const float* __restrict__ y,
                                             float* __restrict__ acc) {
    __shared__ float red[8];
    const size_t total  = (size_t)IMH * IMW;
    size_t i            = (size_t)blockIdx.x * blockDim.x + threadIdx.x;
    const size_t stride = (size_t)gridDim.x * blockDim.x;
    float s = 0.0f;
    for (; i < total; i += stride) {
        float d = x[i] - y[i];
        s += d * d;
    }
    #pragma unroll
    for (int off = 16; off > 0; off >>= 1) s += __shfl_down(s, off, 32);
    if ((threadIdx.x & 31) == 0) red[threadIdx.x >> 5] = s;
    __syncthreads();
    if (threadIdx.x < 8) {
        float v = red[threadIdx.x];
        #pragma unroll
        for (int off = 4; off > 0; off >>= 1) v += __shfl_down(v, off, 8);
        if (threadIdx.x == 0) atomicAdd(acc, v);
    }
}

// ---------------------------------------------------------------------------
// Kernel 2: fused quantize + Sobel(replicate border) + L1 mag + sector NMS +
// double threshold. Emits state byte: bit0 = weak candidate, bit1 = strong.
// One 16x16 tile per block (z selects image), halo 2 staged in LDS.
// ---------------------------------------------------------------------------
__global__ __launch_bounds__(256) void k_front(const float* __restrict__ x,
                                               const float* __restrict__ y,
                                               unsigned char* __restrict__ s0,
                                               unsigned char* __restrict__ s1) {
    const float* img = blockIdx.z ? y : x;
    unsigned char* st = blockIdx.z ? s1 : s0;

    __shared__ float q[20][21];   // quantized pixels, halo 2
    __shared__ float mg[18][19];  // |dx|+|dy|, halo 1

    const int tid = threadIdx.x;
    const int bx = blockIdx.x * 16, by = blockIdx.y * 16;

    for (int i = tid; i < 20 * 20; i += 256) {
        int ly = i / 20, lx = i % 20;
        int gy = min(max(by + ly - 2, 0), IMH - 1);
        int gx = min(max(bx + lx - 2, 0), IMW - 1);
        float v = img[(size_t)gy * IMW + gx];
        q[ly][lx] = floorf(fminf(fmaxf(v, 0.0f), 255.0f));  // uint8 quantize
    }
    __syncthreads();

    for (int i = tid; i < 18 * 18; i += 256) {
        int ly = i / 18 + 1, lx = i % 18 + 1;
        float dx = (q[ly-1][lx+1] - q[ly-1][lx-1]) + 2.0f * (q[ly][lx+1] - q[ly][lx-1])
                 + (q[ly+1][lx+1] - q[ly+1][lx-1]);
        float dy = (q[ly+1][lx-1] - q[ly-1][lx-1]) + 2.0f * (q[ly+1][lx] - q[ly-1][lx])
                 + (q[ly+1][lx+1] - q[ly-1][lx+1]);
        mg[ly-1][lx-1] = fabsf(dx) + fabsf(dy);
    }
    __syncthreads();

    const int tx = tid & 15, ty = tid >> 4;
    const int gx = bx + tx, gy = by + ty;
    const int lyq = ty + 2, lxq = tx + 2;  // center in q coords
    float dx = (q[lyq-1][lxq+1] - q[lyq-1][lxq-1]) + 2.0f * (q[lyq][lxq+1] - q[lyq][lxq-1])
             + (q[lyq+1][lxq+1] - q[lyq+1][lxq-1]);
    float dy = (q[lyq+1][lxq-1] - q[lyq-1][lxq-1]) + 2.0f * (q[lyq+1][lxq] - q[lyq-1][lxq])
             + (q[lyq+1][lxq+1] - q[lyq-1][lxq+1]);
    const int my = ty + 1, mx = tx + 1;    // center in mg coords
    float m   = mg[my][mx];
    float adx = fabsf(dx), ady = fabsf(dy);
    bool horiz = ady <= adx * TG22F;
    bool vert  = ady * TG22F > adx;
    bool same  = dx * dy >= 0.0f;
    float n1 = horiz ? mg[my][mx-1] : (vert ? mg[my-1][mx] : (same ? mg[my-1][mx-1] : mg[my-1][mx+1]));
    float n2 = horiz ? mg[my][mx+1] : (vert ? mg[my+1][mx] : (same ? mg[my+1][mx+1] : mg[my+1][mx-1]));
    bool interior = (gx >= 1) && (gx <= IMW - 2) && (gy >= 1) && (gy <= IMH - 2);
    bool keep = (m > n1) && (m >= n2) && interior;
    unsigned char s = 0;
    if (keep && m > 50.0f) s = (m > 150.0f) ? 3 : 1;  // 3=strong(edge), 1=weak
    st[(size_t)gy * IMW + gx] = s;
}

// ---------------------------------------------------------------------------
// Kernel 3: hysteresis ping-pong. 64x64 tile + halo 1 in LDS; chaotic in-place
// monotone dilation to tile-local fixpoint (unique least fixpoint =>
// deterministic), then write full tile to the *other* buffer (no inter-block
// races across launches).
// ---------------------------------------------------------------------------
__global__ __launch_bounds__(256) void k_hyst(const unsigned char* __restrict__ i0,
                                              const unsigned char* __restrict__ i1,
                                              unsigned char* __restrict__ o0,
                                              unsigned char* __restrict__ o1) {
    const unsigned char* sin = blockIdx.z ? i1 : i0;
    unsigned char* sout      = blockIdx.z ? o1 : o0;

    __shared__ unsigned char st[66][68];
    const int tid = threadIdx.x;
    const int bx = blockIdx.x * 64, by = blockIdx.y * 64;

    for (int i = tid; i < 66 * 66; i += 256) {
        int ly = i / 66, lx = i % 66;
        int gy = by + ly - 1, gx = bx + lx - 1;
        unsigned char v = 0;
        if (gx >= 0 && gx < IMW && gy >= 0 && gy < IMH) v = sin[(size_t)gy * IMW + gx];
        st[ly][lx] = v;
    }
    __syncthreads();

    for (int it = 0; it < 160; ++it) {
        int changed = 0;
        for (int i = tid; i < 64 * 64; i += 256) {
            int ly = (i >> 6) + 1, lx = (i & 63) + 1;
            if (st[ly][lx] == 1) {
                unsigned nb = st[ly-1][lx-1] | st[ly-1][lx] | st[ly-1][lx+1]
                            | st[ly  ][lx-1]               | st[ly  ][lx+1]
                            | st[ly+1][lx-1] | st[ly+1][lx] | st[ly+1][lx+1];
                if (nb & 2u) { st[ly][lx] = 3; changed = 1; }
            }
        }
        if (!__syncthreads_or(changed)) break;
    }

    for (int i = tid; i < 64 * 64; i += 256) {
        int ly = (i >> 6) + 1, lx = (i & 63) + 1;
        sout[(size_t)(by + ly - 1) * IMW + (bx + lx - 1)] = st[ly][lx];
    }
}

// ---------------------------------------------------------------------------
// Kernel 4: |e1 - e2| as f16, stored TRANSPOSED (column-major -> row-major)
// so the WMMA B-matrix feed becomes contiguous 32B loads per lane.
// ---------------------------------------------------------------------------
__global__ __launch_bounds__(256) void k_difftr(const unsigned char* __restrict__ s0,
                                                const unsigned char* __restrict__ s1,
                                                _Float16* __restrict__ dT) {
    __shared__ _Float16 t[64][65];
    const int tid = threadIdx.x;
    const int bx = blockIdx.x * 64;  // column base
    const int by = blockIdx.y * 64;  // row base
    for (int i = tid; i < 64 * 64; i += 256) {
        int ly = i >> 6, lx = i & 63;  // ly=row offs, lx=col offs (coalesced)
        size_t idx = (size_t)(by + ly) * IMW + (bx + lx);
        int d = ((s0[idx] >> 1) ^ (s1[idx] >> 1)) & 1;  // |e1-e2| for binary
        t[lx][ly] = (_Float16)d;
    }
    __syncthreads();
    for (int i = tid; i < 64 * 64; i += 256) {
        int ln = i >> 6, lk = i & 63;  // ln=col(n), lk=row(k); contiguous in k
        dT[(size_t)(bx + ln) * IMH + (by + lk)] = t[ln][lk];
    }
}

// ---------------------------------------------------------------------------
// Kernel 5: column sums via v_wmma_f32_16x16x32_f16 with A = ones.
// D[m][n] = sum_k B[k][n]  (every output row equals the column sums; exact
// since diffs are {0,1} and K=4096 < 2^24). One wave per 16-column group.
// B k order used here (wave32): lane = N (lanes 0-15: K 0..15, lanes 16-31: K 16..31),
// each lane's 16 halves contiguous in the transposed buffer -> 32B loads.
// ---------------------------------------------------------------------------
__global__ __launch_bounds__(256) void k_colsum(const _Float16* __restrict__ dT,
                                                float* __restrict__ cs) {
    const int wave  = threadIdx.x >> 5;
    const int lane  = threadIdx.x & 31;
    const int group = blockIdx.x * 8 + wave;
    const int n     = (group << 4) + (lane & 15);
    const int koff  = (lane >> 4) << 4;

    v16h a;
    #pragma unroll
    for (int i = 0; i < 16; ++i) a[i] = (_Float16)1.0f;

    v8f acc0 = {};
    v8f acc1 = {};
    const _Float16* base = dT + (size_t)n * IMH + koff;
    #pragma unroll 4
    for (int kb = 0; kb < IMH; kb += 64) {
        v16h b0 = *(const v16h*)(base + kb);
        v16h b1 = *(const v16h*)(base + kb + 32);
        acc0 = __builtin_amdgcn_wmma_f32_16x16x32_f16(false, a, false, b0,
                                                      (short)0, acc0, false, false);
        acc1 = __builtin_amdgcn_wmma_f32_16x16x32_f16(false, a, false, b1,
                                                      (short)0, acc1, false, false);
    }
    float r = acc0[0] + acc1[0];  // row M=0 (lanes 0-15) / M=8 (16-31): identical
    if (lane < 16) cs[(group << 4) + lane] = r;
}

// ---------------------------------------------------------------------------
// Kernel 6: out[n] = colsum[n] + mse_sum / (H*W)
// ---------------------------------------------------------------------------
__global__ void k_finalize(const float* __restrict__ cs,
                           const float* __restrict__ acc,
                           float* __restrict__ out) {
    int n = blockIdx.x * blockDim.x + threadIdx.x;
    if (n < IMW) out[n] = cs[n] + (*acc) * (1.0f / ((float)IMH * (float)IMW));
}

// ---------------------------------------------------------------------------
extern "C" void kernel_launch(void* const* d_in, const int* in_sizes, int n_in,
                              void* d_out, int out_size, void* d_ws, size_t ws_size,
                              hipStream_t stream) {
    (void)in_sizes; (void)n_in; (void)out_size; (void)ws_size;
    const float* x = (const float*)d_in[0];
    const float* y = (const float*)d_in[1];
    float* out = (float*)d_out;

    char* ws = (char*)d_ws;
    const size_t NPIX = (size_t)IMH * IMW;
    unsigned char* sA0 = (unsigned char*)ws;            // state img0, buf A
    unsigned char* sA1 = sA0 + NPIX;                    // state img1, buf A
    unsigned char* sB0 = sA1 + NPIX;                    // state img0, buf B
    unsigned char* sB1 = sB0 + NPIX;                    // state img1, buf B
    _Float16* dT = (_Float16*)sB0;                      // aliases B (dead after even #iters)
    float* cs  = (float*)(ws + 4 * NPIX);               // 4096 column sums
    float* acc = cs + IMW;                              // MSE accumulator

    k_zero<<<1, 32, 0, stream>>>(acc);
    k_mse<<<2048, 256, 0, stream>>>(x, y, acc);
    k_front<<<dim3(IMW / 16, IMH / 16, 2), 256, 0, stream>>>(x, y, sA0, sA1);

    // 6 ping-pong hysteresis passes: >= 6*64 >= 256 propagation steps (ref cap).
    // Even count => final state lands back in buffer A.
    for (int it = 0; it < 6; ++it) {
        const unsigned char* i0 = (it & 1) ? sB0 : sA0;
        const unsigned char* i1 = (it & 1) ? sB1 : sA1;
        unsigned char* o0 = (it & 1) ? sA0 : sB0;
        unsigned char* o1 = (it & 1) ? sA1 : sB1;
        k_hyst<<<dim3(IMW / 64, IMH / 64, 2), 256, 0, stream>>>(i0, i1, o0, o1);
    }

    k_difftr<<<dim3(IMW / 64, IMH / 64, 1), 256, 0, stream>>>(sA0, sA1, dT);
    k_colsum<<<IMW / 16 / 8, 256, 0, stream>>>(dT, cs);
    k_finalize<<<(IMW + 255) / 256, 256, 0, stream>>>(cs, acc, out);
}
